// SlimTritonINT4Linear_71468255805756
// MI455X (gfx1250) — matrix operand order repaired and checked by reading, not compiled, not run on hardware
//
#include <hip/hip_runtime.h>
#include <stdint.h>

typedef _Float16 v16h __attribute__((ext_vector_type(16)));
typedef _Float16 v8h  __attribute__((ext_vector_type(8)));
typedef _Float16 v2h  __attribute__((ext_vector_type(2)));
typedef float    v8f  __attribute__((ext_vector_type(8)));

constexpr int IN_F  = 4096;
constexpr int OUT_F = 11008;
constexpr int GROUP = 128;
constexpr int M_TOK = 128;

constexpr int BM = 128;    // block covers ALL token rows -> weights touched once
constexpr int BN = 64;     // cols per block
constexpr int BK = GROUP;  // K chunk == one quant group
constexpr int NCHUNK = IN_F / BK;  // 32

// LDS row pitch: 136 halves = 272 B = 17 * 16 B -> 16B-aligned segments,
// rows staggered across the 64 banks.
constexpr int XP = 136;
constexpr int WP = 136;

__global__ __launch_bounds__(256) void w4a16_gemm_wmma(
    const _Float16* __restrict__ x,        // [128, 4096] fp16
    const int*      __restrict__ qweight,  // [2048, 11008] two nibbles/elem
    const _Float16* __restrict__ scales,   // [32, 11008] fp16
    const int*      __restrict__ qzeros,   // [16, 11008] two nibbles/elem
    _Float16*       __restrict__ out)      // [128, 11008] fp16
{
  __shared__ _Float16 xs[BM][XP];   // activation chunk (async-copied)
  __shared__ _Float16 ws[BN][WP];   // dequantized weight chunk

  const int tid  = threadIdx.x;
  const int lane = tid & 31;        // wave32 lane
  const int wv   = tid >> 5;        // wave id 0..7
  const int half = lane >> 4;       // 0/1 lane group
  const int l16  = lane & 15;

  const int n0 = blockIdx.x * BN;

  // 8 waves = 4 (M) x 2 (N); each wave owns a 32x32 register tile
  const int wm = wv >> 1;
  const int wn = wv & 1;
  const int lmA = wm * 32 + l16;    // A fragment rows (tile 0; tile 1 = +16)
  const int lnB = wn * 32 + l16;    // B fragment rows (tile 0; tile 1 = +16)

  // dequant assignment: thread -> fixed column, 4-strided packed rows
  const int nl    = tid & 63;       // 0..63
  const int nglob = n0 + nl;
  const int kpb   = tid >> 6;       // 0..3

  v8f acc[2][2] = {{{}, {}}, {{}, {}}};

  // ---- software pipeline: packed weights for chunk 0 into registers -------
  int pk[16];
  {
    const int* q0 = qweight + nglob;
    #pragma unroll
    for (int it = 0; it < 16; ++it)
      pk[it] = q0[(size_t)(kpb + it * 4) * OUT_F];
  }

  for (int g = 0; g < NCHUNK; ++g) {
    const int kc = g * BK;

    // ---- async-copy this x chunk into LDS (wave wv owns rows wv*16..+15);
    //      transfers overlap the dequant VALU below
    #pragma unroll
    for (int i = 0; i < 8; ++i) {
      const int linear = i * 32 + lane;    // 0..255: 16 rows x 16 segs (16B)
      const int r  = linear >> 4;
      const int s  = linear & 15;
      const int lm = wv * 16 + r;
      const _Float16* src = x + (size_t)lm * IN_F + kc + s * 8;
      unsigned           lds = (unsigned)(uintptr_t)&xs[lm][s * 8];
      unsigned long long ga  = (unsigned long long)(uintptr_t)src;
      asm volatile("global_load_async_to_lds_b128 %0, %1, off"
                   :: "v"(lds), "v"(ga) : "memory");
    }

    // ---- dequant 128x64 weight tile from the pre-loaded registers.
    //      fp16 magic: 0x640n == 1024+n, so per packed byte we need only
    //      3 bit-ops + one v_pk_fma_f16 with w = (1024+n)*s - (1024+z)*s.
    {
      const float scalef = (float)scales[(size_t)g * OUT_F + nglob];
      const int   zq     = qzeros[(size_t)(g >> 1) * OUT_F + nglob];
      const int   zero   = (zq >> ((g & 1) * 4)) & 15;
      const _Float16 sh  = (_Float16)scalef;
      const _Float16 bh  = (_Float16)(-scalef * (float)(1024 + zero));
      v2h s2; s2.x = sh; s2.y = sh;
      v2h b2; b2.x = bh; b2.y = bh;
      #pragma unroll
      for (int it = 0; it < 16; ++it) {
        const int kp     = kpb + it * 4;          // packed row 0..63
        const int packed = pk[it];
        const int t = ((packed & 15) | ((packed << 12) & 0xF0000)) | 0x64006400;
        const v2h h2 = __builtin_bit_cast(v2h, t);    // {1024+lo, 1024+hi}
        const v2h wp = h2 * s2 + b2;                  // v_pk_fma_f16
        *(v2h*)&ws[nl][2 * kp] = wp;
      }
    }

    // ---- issue next chunk's packed-weight loads; with the hand-rolled
    //      barrier below (no LOADcnt drain) they stay in flight across the
    //      entire WMMA phase
    {
      const int  gn = (g + 1 < NCHUNK) ? g + 1 : NCHUNK - 1;
      const int* qn = qweight + (size_t)(gn * (BK / 2)) * OUT_F + nglob;
      #pragma unroll
      for (int it = 0; it < 16; ++it)
        pk[it] = qn[(size_t)(kpb + it * 4) * OUT_F];
    }

    // ---- prefetch the g+2 weight tile into L2 (256 threads cover all
    //      64 rows x 4 x 64B segments of this block's column stripe)
    {
      const int gp = (g + 2 < NCHUNK) ? g + 2 : NCHUNK - 1;
      const int* pf = qweight + (size_t)(gp * (BK / 2) + (tid & 63)) * OUT_F
                      + n0 + (tid >> 6) * 16;
      __builtin_prefetch(pf, 0, 0);
    }

    // ---- split barrier: wait ONLY async (x tile) + LDS (ws stores); the 16
    //      outstanding global b32 loads keep streaming through the barrier
    asm volatile(
        "s_wait_asynccnt 0\n\t"
        "s_wait_dscnt 0\n\t"
        "s_barrier_signal -1\n\t"
        "s_barrier_wait -1"
        ::: "memory");

    // ---- 16 WMMAs over this K chunk: 32x32 tile, 2:1 ds:wmma ratio --------
    #pragma unroll
    for (int ks = 0; ks < 4; ++ks) {
      const int koff = ks * 32;
      // A 16x32 fp16: VGPR0-3 <- K = half*8+0..7, VGPR4-7 <- K = 16+half*8+0..7
      const v8h a00 = *(const v8h*)&xs[lmA     ][koff + half * 8];
      const v8h a01 = *(const v8h*)&xs[lmA     ][koff + 16 + half * 8];
      const v8h a10 = *(const v8h*)&xs[lmA + 16][koff + half * 8];
      const v8h a11 = *(const v8h*)&xs[lmA + 16][koff + 16 + half * 8];
      // B 32x16 fp16: VGPR0-3 <- K = half*8+0..7, VGPR4-7 <- K = 16+half*8+0..7
      const v8h b00 = *(const v8h*)&ws[lnB     ][koff + half * 8];
      const v8h b01 = *(const v8h*)&ws[lnB     ][koff + 16 + half * 8];
      const v8h b10 = *(const v8h*)&ws[lnB + 16][koff + half * 8];
      const v8h b11 = *(const v8h*)&ws[lnB + 16][koff + 16 + half * 8];
      const v16h a0 = __builtin_shufflevector(a00, a01,
          0,1,2,3,4,5,6,7,8,9,10,11,12,13,14,15);
      const v16h a1 = __builtin_shufflevector(a10, a11,
          0,1,2,3,4,5,6,7,8,9,10,11,12,13,14,15);
      const v16h b0 = __builtin_shufflevector(b00, b01,
          0,1,2,3,4,5,6,7,8,9,10,11,12,13,14,15);
      const v16h b1 = __builtin_shufflevector(b10, b11,
          0,1,2,3,4,5,6,7,8,9,10,11,12,13,14,15);
      acc[0][0] = __builtin_amdgcn_wmma_f32_16x16x32_f16(
          false, a0, false, b0, (short)0, acc[0][0], false, false);
      acc[0][1] = __builtin_amdgcn_wmma_f32_16x16x32_f16(
          false, a0, false, b1, (short)0, acc[0][1], false, false);
      acc[1][0] = __builtin_amdgcn_wmma_f32_16x16x32_f16(
          false, a1, false, b0, (short)0, acc[1][0], false, false);
      acc[1][1] = __builtin_amdgcn_wmma_f32_16x16x32_f16(
          false, a1, false, b1, (short)0, acc[1][1], false, false);
    }

    // ---- tail barrier: fragment ds_loads are complete (dscnt waited by the
    //      WMMA feeds); protect xs/ws before the next chunk overwrites them
    asm volatile(
        "s_wait_dscnt 0\n\t"
        "s_barrier_signal -1\n\t"
        "s_barrier_wait -1"
        ::: "memory");
  }

  // ---- epilogue: f32 C 16x16 layout -> VGPR j holds row (half*8 + j) ------
  #pragma unroll
  for (int mt = 0; mt < 2; ++mt) {
    #pragma unroll
    for (int nt = 0; nt < 2; ++nt) {
      const int cn = n0 + wn * 32 + nt * 16 + l16;
      #pragma unroll
      for (int j = 0; j < 8; ++j) {
        const int cm = wm * 32 + mt * 16 + half * 8 + j;
        out[(size_t)cm * OUT_F + cn] = (_Float16)acc[mt][nt][j];
      }
    }
  }
}

extern "C" void kernel_launch(void* const* d_in, const int* in_sizes, int n_in,
                              void* d_out, int out_size, void* d_ws, size_t ws_size,
                              hipStream_t stream) {
  (void)in_sizes; (void)n_in; (void)out_size; (void)d_ws; (void)ws_size;
  const _Float16* x       = (const _Float16*)d_in[0];
  const int*      qweight = (const int*)d_in[1];
  const _Float16* scales  = (const _Float16*)d_in[2];
  const int*      qzeros  = (const int*)d_in[3];
  _Float16*       out     = (_Float16*)d_out;

  dim3 grid(OUT_F / BN);   // 172 blocks, each owns a 64-col stripe, all of M
  dim3 block(256);         // 8 wave32 waves
  hipLaunchKernelGGL(w4a16_gemm_wmma, grid, block, 0, stream,
                     x, qweight, scales, qzeros, out);
}
